// CustomConvolution2d_51788715655282
// MI455X (gfx1250) — matrix operand order repaired and checked by reading, not compiled, not run on hardware
//
#include <hip/hip_runtime.h>

typedef __attribute__((ext_vector_type(16))) __bf16 v16bf;
typedef __attribute__((ext_vector_type(8)))  float  v8f;
typedef __attribute__((ext_vector_type(4)))  unsigned int u32x4;

#define C_IN   64
#define K_OUT  64
#define H_DIM  128
#define W_DIM  128
#define N_BAT  32
#define KTOT   576           // C_IN * 9
#define ROWS   4             // staged padded input rows per block (h0-1 .. h0+2)
#define PSTR   72            // padded c-stride (bf16 elems) in patch LDS
#define WSTR   584           // padded k-stride (bf16 elems) in weight LDS
#define PS_ELEMS (ROWS * 130 * PSTR)
#define WL_ELEMS (K_OUT * WSTR)
#define LDS_BYTES (PS_ELEMS * 2 + WL_ELEMS * 2 + K_OUT * 4)

__device__ __forceinline__ __bf16 f2bf(float f) { return (__bf16)f; }

extern "C" __global__ __launch_bounds__(256)
void conv3x3_wmma_bf16(const float* __restrict__ x,
                       const float* __restrict__ wgt,
                       const float* __restrict__ bias,
                       float* __restrict__ out)
{
    extern __shared__ char lds_raw[];
    __bf16* ps   = (__bf16*)lds_raw;                       // [ROWS][130][PSTR] patches
    __bf16* wlds = (__bf16*)(lds_raw + PS_ELEMS * 2);      // [64][WSTR] weights, k=(rs)*64+c
    float*  blds = (float*) (lds_raw + PS_ELEMS * 2 + WL_ELEMS * 2);

    const int tid = threadIdx.x;
    const int blk = blockIdx.x;
    const int n   = blk >> 6;          // image
    const int h0  = (blk & 63) << 1;   // first of two output rows

    // ---- stage weights (coalesced read, LDS transpose k: c*9+rs -> rs*64+c) ----
    for (int idx = tid; idx < K_OUT * KTOT; idx += 256) {
        int ko = idx / KTOT;
        int t  = idx - ko * KTOT;      // t = c*9 + rs  (contiguous in OIHW)
        int c  = t / 9;
        int rs = t - c * 9;
        wlds[ko * WSTR + rs * 64 + c] = f2bf(wgt[idx]);
    }
    if (tid < K_OUT) blds[tid] = bias[tid];

    // ---- stage ROWS padded input rows as ps[rowIdx][w'][c] (coalesced along w') ----
    const float* xn = x + (long)n * (C_IN * H_DIM * W_DIM);
    for (int idx = tid; idx < C_IN * ROWS * 130; idx += 256) {
        int wp = idx % 130;            // w' = w + 1 (padded)
        int cr = idx / 130;
        int r4 = cr % ROWS;            // rowIdx: global row h0 - 1 + r4
        int c  = cr / ROWS;
        int gh = h0 + r4 - 1;
        int gw = wp - 1;
        float v = 0.0f;
        if ((unsigned)gh < H_DIM && (unsigned)gw < W_DIM)
            v = xn[(c * H_DIM + gh) * W_DIM + gw];
        ps[(r4 * 130 + wp) * PSTR + c] = f2bf(v);
    }
    __syncthreads();

    // ---- wave tiling: wave = 32(M) x 64(N) of the 64(M) x 256(N) block tile ----
    const int lane = tid & 31;
    const int wv   = tid >> 5;         // 0..7
    const int lid  = lane & 15;
    const int half = lane >> 4;        // 0/1
    const int mg   = wv & 1;           // M group: ko base = mg*32, tiles +0/+16
    const int ng   = wv >> 1;          // N group 0..3
    const int kobase = mg << 5;
    const int sel    = ng >> 1;        // which of the two output rows (0/1)
    const int wbase  = (ng & 1) << 6;  // w base within the row (0/64)

    v8f acc00 = {}, acc01 = {}, acc02 = {}, acc03 = {};
    v8f acc10 = {}, acc11 = {}, acc12 = {}, acc13 = {};

    // Loop-invariant base pointers; per-chunk deltas are compile-time constants
    // after full unroll, so every ds_load gets an immediate offset.
    const __bf16* abase0 = wlds + (kobase + lid) * WSTR + (half << 4);
    const __bf16* abase1 = wlds + (kobase + 16 + lid) * WSTR + (half << 4);
    const __bf16* bbase0 = ps + (sel * 130 + wbase + 0  + lid) * PSTR + (half << 4);
    const __bf16* bbase1 = ps + (sel * 130 + wbase + 16 + lid) * PSTR + (half << 4);
    const __bf16* bbase2 = ps + (sel * 130 + wbase + 32 + lid) * PSTR + (half << 4);
    const __bf16* bbase3 = ps + (sel * 130 + wbase + 48 + lid) * PSTR + (half << 4);

    #pragma unroll
    for (int kc = 0; kc < 18; ++kc) {
        const int rs = kc >> 1;
        const int cb = (kc & 1) << 5;  // channel base 0 / 32
        const int r  = rs / 3;
        const int s  = rs - r * 3;
        const int k0 = kc << 5;
        const int boff = (r * 130 + s) * PSTR + cb;   // compile-time per iteration

        union { v16bf v; u32x4 q[2]; } A0, A1;
        A0.q[0] = *(const u32x4*)(abase0 + k0);
        A0.q[1] = *(const u32x4*)(abase0 + k0 + 8);
        A1.q[0] = *(const u32x4*)(abase1 + k0);
        A1.q[1] = *(const u32x4*)(abase1 + k0 + 8);

        union { v16bf v; u32x4 q[2]; } B0, B1, B2, B3;
        B0.q[0] = *(const u32x4*)(bbase0 + boff);  B0.q[1] = *(const u32x4*)(bbase0 + boff + 8);
        B1.q[0] = *(const u32x4*)(bbase1 + boff);  B1.q[1] = *(const u32x4*)(bbase1 + boff + 8);
        B2.q[0] = *(const u32x4*)(bbase2 + boff);  B2.q[1] = *(const u32x4*)(bbase2 + boff + 8);
        B3.q[0] = *(const u32x4*)(bbase3 + boff);  B3.q[1] = *(const u32x4*)(bbase3 + boff + 8);

        acc00 = __builtin_amdgcn_wmma_f32_16x16x32_bf16(false, A0.v, false, B0.v, (short)0, acc00, false, false);
        acc01 = __builtin_amdgcn_wmma_f32_16x16x32_bf16(false, A0.v, false, B1.v, (short)0, acc01, false, false);
        acc02 = __builtin_amdgcn_wmma_f32_16x16x32_bf16(false, A0.v, false, B2.v, (short)0, acc02, false, false);
        acc03 = __builtin_amdgcn_wmma_f32_16x16x32_bf16(false, A0.v, false, B3.v, (short)0, acc03, false, false);
        acc10 = __builtin_amdgcn_wmma_f32_16x16x32_bf16(false, A1.v, false, B0.v, (short)0, acc10, false, false);
        acc11 = __builtin_amdgcn_wmma_f32_16x16x32_bf16(false, A1.v, false, B1.v, (short)0, acc11, false, false);
        acc12 = __builtin_amdgcn_wmma_f32_16x16x32_bf16(false, A1.v, false, B2.v, (short)0, acc12, false, false);
        acc13 = __builtin_amdgcn_wmma_f32_16x16x32_bf16(false, A1.v, false, B3.v, (short)0, acc13, false, false);
    }

    // ---- epilogue: C/D layout VGPR i -> row (i + half*8), col = lid ----
    const int hrow = h0 + sel;
    float* outn = out + ((long)n * K_OUT) * (H_DIM * W_DIM) + hrow * W_DIM;
    #pragma unroll
    for (int i = 0; i < 8; ++i) {
        const int ko0 = kobase + (half << 3) + i;
        const int ko1 = ko0 + 16;
        const float b0 = blds[ko0];
        const float b1 = blds[ko1];
        float* orow0 = outn + ko0 * (H_DIM * W_DIM);
        float* orow1 = outn + ko1 * (H_DIM * W_DIM);
        orow0[wbase + 0  + lid] = acc00[i] + b0;
        orow0[wbase + 16 + lid] = acc01[i] + b0;
        orow0[wbase + 32 + lid] = acc02[i] + b0;
        orow0[wbase + 48 + lid] = acc03[i] + b0;
        orow1[wbase + 0  + lid] = acc10[i] + b1;
        orow1[wbase + 16 + lid] = acc11[i] + b1;
        orow1[wbase + 32 + lid] = acc12[i] + b1;
        orow1[wbase + 48 + lid] = acc13[i] + b1;
    }
}

extern "C" void kernel_launch(void* const* d_in, const int* in_sizes, int n_in,
                              void* d_out, int out_size, void* d_ws, size_t ws_size,
                              hipStream_t stream) {
    (void)in_sizes; (void)n_in; (void)out_size; (void)d_ws; (void)ws_size;
    const float* x    = (const float*)d_in[0];
    const float* wgt  = (const float*)d_in[1];
    const float* bias = (const float*)d_in[2];
    float* out        = (float*)d_out;

    dim3 grid(N_BAT * (H_DIM / 2));  // one (n, h0..h0+1) row pair per workgroup
    dim3 block(256);                 // 8 wave32
    hipLaunchKernelGGL(conv3x3_wmma_bf16, grid, block, LDS_BYTES, stream,
                       x, wgt, bias, out);
}
